// HeteroGNNLayer_10050223473001
// MI455X (gfx1250) — matrix operand order repaired and checked by reading, not compiled, not run on hardware
//
#include <hip/hip_runtime.h>
#include <math.h>

typedef __bf16 v16bf __attribute__((ext_vector_type(16)));
typedef float  v8f   __attribute__((ext_vector_type(8)));

#define CC 256
#define SCALE 0.25f   // HD^-0.5, HD=16
#define EPS 1e-5f

// Fragment register image: 16 bf16 = 8 dwords = 32 bytes.
union AFrag {
    v16bf    v;
    unsigned u[8];
    __bf16   h[16];
};

// ---------------------------------------------------------------------------
// One-time weight prep: dst[c*rows + r] = bf16(src[r*cols + c])
// (i.e. store W^T in bf16 so hot loops read bf16 B elements without conversion)
// ---------------------------------------------------------------------------
__global__ __launch_bounds__(256) void transpose_cvt(
    const float* __restrict__ src, __bf16* __restrict__ dst, int rows, int cols)
{
    int i = blockIdx.x * 256 + threadIdx.x;
    if (i < rows * cols) {
        int r = i / cols, c = i - r * cols;
        dst[c * rows + r] = (__bf16)src[i];
    }
}

// ---------------------------------------------------------------------------
// Y[m][n] = sum_k X[m][k]*WT[k][n] + bias[n]   (K = N = 256)
// Block: 256 threads = 8 waves; block tile 32 rows x 256 cols.
// Wave w: rows 16*(w>>2), cols 64*(w&3) -> 4 WMMA col tiles, 8 K-steps of 32.
// WT chunk (32x256 bf16, already transposed) staged into LDS with b128 copies.
// OUT_BF16: emit packed bf16 (xp buffers). FUSE_LN: LayerNorm epilogue via LDS.
// ---------------------------------------------------------------------------
template<bool OUT_BF16, bool FUSE_LN>
__global__ __launch_bounds__(256) void gemm256(
    const float* __restrict__ X, const __bf16* __restrict__ WT,
    const float* __restrict__ bias,
    float* __restrict__ Yf, __bf16* __restrict__ Y16,
    const float* __restrict__ lng, const float* __restrict__ lnb, int nrows)
{
    __shared__ __bf16 Bs[32 * CC];    // 16KB: straight copy of WT rows k0..k0+31
    __shared__ float  Zs[32 * CC];    // 32KB: LN staging (FUSE_LN only)

    const int tid  = threadIdx.x;
    const int lane = tid & 31;
    const int w    = tid >> 5;
    const int row0 = blockIdx.x * 32 + (w >> 2) * 16;
    const int col0 = (w & 3) * 64;
    const int arow = row0 + (lane & 15);
    const int khalf = (lane >> 4) ? 8 : 0;
    const int bk    = (lane & 15) + ((lane >> 4) << 4);

    v8f acc[4] = {};

    for (int kc = 0; kc < 8; ++kc) {
        const int k0 = kc * 32;
        // stage B chunk: 16KB contiguous bf16, b128 copies
        {
            const uint4* s4 = (const uint4*)(WT + (size_t)k0 * CC);
            uint4* d4 = (uint4*)Bs;
            for (int i = tid; i < 1024; i += 256) d4[i] = s4[i];
        }
        __syncthreads();

        const float* xr = X + (size_t)arow * CC + k0;
        if (kc < 7) __builtin_prefetch(xr + 32, 0, 1);

        AFrag a;
        #pragma unroll
        for (int v = 0; v < 8; ++v) {
            int kk = ((v < 4) ? (2 * v) : (2 * v - 8 + 16)) + khalf;
            a.h[2 * v]     = (__bf16)xr[kk];
            a.h[2 * v + 1] = (__bf16)xr[kk + 1];
        }
        #pragma unroll
        for (int ct = 0; ct < 4; ++ct) {
            const uint4* bs4 = (const uint4*)(Bs + bk * CC + col0 + ct * 16);
            AFrag b;
            #pragma unroll
            for (int s = 0; s < 16; ++s)
                b.h[s] = Bs[(((s < 8) ? s : s + 8) + khalf) * CC + col0 + ct * 16 + (lane & 15)];
            acc[ct] = __builtin_amdgcn_wmma_f32_16x16x32_bf16(
                false, a.v, false, b.v, (short)0, acc[ct], false, false);
        }
        __syncthreads();
    }

    if constexpr (OUT_BF16) {
        #pragma unroll
        for (int ct = 0; ct < 4; ++ct) {
            #pragma unroll
            for (int j = 0; j < 8; ++j) {
                int r = row0 + ((lane < 16) ? j : j + 8);
                int c = col0 + ct * 16 + (lane & 15);
                if (r < nrows) Y16[(size_t)r * CC + c] = (__bf16)(acc[ct][j] + bias[c]);
            }
        }
    } else if constexpr (FUSE_LN) {
        #pragma unroll
        for (int ct = 0; ct < 4; ++ct) {
            #pragma unroll
            for (int j = 0; j < 8; ++j) {
                int rl = (w >> 2) * 16 + ((lane < 16) ? j : j + 8);
                int c  = col0 + ct * 16 + (lane & 15);
                Zs[rl * CC + c] = acc[ct][j] + bias[c];
            }
        }
        __syncthreads();
        #pragma unroll
        for (int i = 0; i < 4; ++i) {
            const int rl  = w * 4 + i;
            const int row = blockIdx.x * 32 + rl;
            float vv[8], s = 0.f;
            #pragma unroll
            for (int t = 0; t < 8; ++t) { vv[t] = Zs[rl * CC + lane + t * 32]; s += vv[t]; }
            #pragma unroll
            for (int off = 16; off; off >>= 1) s += __shfl_xor(s, off, 32);
            const float mean = s * (1.f / 256.f);
            float vs = 0.f;
            #pragma unroll
            for (int t = 0; t < 8; ++t) { float d = vv[t] - mean; vs += d * d; }
            #pragma unroll
            for (int off = 16; off; off >>= 1) vs += __shfl_xor(vs, off, 32);
            const float inv = rsqrtf(vs * (1.f / 256.f) + EPS);
            if (row < nrows) {
                #pragma unroll
                for (int t = 0; t < 8; ++t) {
                    int c = lane + t * 32;
                    Yf[(size_t)row * CC + c] = (vv[t] - mean) * inv * lng[c] + lnb[c];
                }
            }
        }
    } else {
        #pragma unroll
        for (int ct = 0; ct < 4; ++ct) {
            #pragma unroll
            for (int j = 0; j < 8; ++j) {
                int r = row0 + ((lane < 16) ? j : j + 8);
                int c = col0 + ct * 16 + (lane & 15);
                if (r < nrows) Yf[(size_t)r * CC + c] = acc[ct][j] + bias[c];
            }
        }
    }
}

// ---------------------------------------------------------------------------
// Edge-message MHA with fused scatter-add.
// Block: 128 threads = 4 waves; wave owns 4 edges = one 16x64 row tile
// (rows = (edge, head-chunk)). xp inputs are bf16 -> A-frags are raw u32 loads.
// Weights pre-transposed bf16 -> B-frags are 16 strided bf16 loads, no conversion.
// ---------------------------------------------------------------------------
__global__ __launch_bounds__(128) void edge_mha_scatter(
    const __bf16* __restrict__ xp_src, const __bf16* __restrict__ xp_dst,
    const int*    __restrict__ ei,     // [2][E]
    const float*  __restrict__ ea,     // [E]
    const float*  __restrict__ We,     // [256]
    const float*  __restrict__ be,     // [256]
    const __bf16* __restrict__ WTin,   // [64][192] = Win^T (q|k|v side by side)
    const float*  __restrict__ bin,    // [192]
    const __bf16* __restrict__ WTout,  // [64][64]  = Wout^T
    const float*  __restrict__ bout,   // [64]
    float* __restrict__ outbuf,        // [N][256], atomic accumulate at dst
    int E_)
{
    __shared__ float Qs[4][16 * 64];   // reused to hold O after attention
    __shared__ float Ks[4][16 * 64];
    __shared__ float Vs[4][16 * 64];

    const int lane = threadIdx.x & 31;
    const int wid  = threadIdx.x >> 5;
    const int e0   = (blockIdx.x * 4 + wid) * 4;

    const int rr    = lane & 15;
    const int e_l   = rr >> 2;
    const int h_in  = rr & 3;
    const int khalf = (lane >> 4) ? 8 : 0;
    const int bk0   = (lane & 15) + ((lane >> 4) << 4);

    const int   s_idx = ei[e0 + e_l];
    const int   d_idx = ei[(size_t)E_ + e0 + e_l];
    const float eaval = ea[e0 + e_l];

    const unsigned* q32 = (const unsigned*)(xp_dst + (size_t)d_idx * CC + h_in * 64);
    const unsigned* k32 = (const unsigned*)(xp_src + (size_t)s_idx * CC + h_in * 64);

    // ---- projections: p=0 Q(dst), p=1 K(src), p=2 V(edge feature) ----
    for (int p = 0; p < 3; ++p) {
        v8f acc[4] = {};
        #pragma unroll
        for (int kc = 0; kc < 2; ++kc) {
            AFrag a;
            #pragma unroll
            for (int v = 0; v < 8; ++v) {
                int kk = ((v < 4) ? (2 * v) : (2 * v - 8 + 16)) + khalf + kc * 32;
                if (p == 0)      a.u[v] = q32[kk >> 1];
                else if (p == 1) a.u[v] = k32[kk >> 1];
                else {
                    int c0 = h_in * 64 + kk;
                    a.h[2 * v]     = (__bf16)(eaval * We[c0]     + be[c0]);
                    a.h[2 * v + 1] = (__bf16)(eaval * We[c0 + 1] + be[c0 + 1]);
                }
            }
            const int bkk = bk0 + kc * 32;
            #pragma unroll
            for (int ct = 0; ct < 4; ++ct) {
                const uint4* w4 = (const uint4*)(WTin + (size_t)bkk * 192 + p * 64 + ct * 16);
                AFrag b;
                #pragma unroll
                for (int s = 0; s < 16; ++s)
                    b.h[s] = WTin[(size_t)(((s < 8) ? s : s + 8) + khalf + kc * 32) * 192 + p * 64 + ct * 16 + rr];
                acc[ct] = __builtin_amdgcn_wmma_f32_16x16x32_bf16(
                    false, a.v, false, b.v, (short)0, acc[ct], false, false);
            }
        }
        float* dst = (p == 0) ? &Qs[wid][0] : (p == 1) ? &Ks[wid][0] : &Vs[wid][0];
        #pragma unroll
        for (int ct = 0; ct < 4; ++ct) {
            #pragma unroll
            for (int j = 0; j < 8; ++j) {
                int r = (lane < 16) ? j : j + 8;
                int c = ct * 16 + (lane & 15);
                dst[r * 64 + c] = acc[ct][j] + bin[p * 64 + c];
            }
        }
    }
    __builtin_amdgcn_wave_barrier();   // LDS in-order within the wave

    // ---- attention: 2 lanes per (edge,head); each lane handles 2 q rows ----
    {
        const int eh = lane >> 1;
        const int ee = eh >> 2, h = eh & 3;
        const int qsel = (lane & 1) * 2;
        #pragma unroll
        for (int qi = 0; qi < 2; ++qi) {
            const int rq = ee * 4 + (qsel + qi);
            float s[4], mx = -1e30f;
            #pragma unroll
            for (int k = 0; k < 4; ++k) {
                float d = 0.f;
                #pragma unroll
                for (int hd = 0; hd < 16; ++hd)
                    d += Qs[wid][rq * 64 + h * 16 + hd] * Ks[wid][(ee * 4 + k) * 64 + h * 16 + hd];
                s[k] = d * SCALE;
                mx = fmaxf(mx, s[k]);
            }
            float den = 0.f;
            #pragma unroll
            for (int k = 0; k < 4; ++k) { s[k] = __expf(s[k] - mx); den += s[k]; }
            const float inv = 1.f / den;
            #pragma unroll
            for (int hd = 0; hd < 16; ++hd) {
                float o = 0.f;
                #pragma unroll
                for (int k = 0; k < 4; ++k)
                    o += s[k] * Vs[wid][(ee * 4 + k) * 64 + h * 16 + hd];
                Qs[wid][rq * 64 + h * 16 + hd] = o * inv;   // O overwrites Q
            }
        }
    }
    __builtin_amdgcn_wave_barrier();

    // ---- out projection: O(16x64) @ Wout^T + bout, atomic scatter ----
    {
        v8f acc[4] = {};
        #pragma unroll
        for (int kc = 0; kc < 2; ++kc) {
            AFrag a;
            #pragma unroll
            for (int v = 0; v < 8; ++v) {
                int kk = ((v < 4) ? (2 * v) : (2 * v - 8 + 16)) + khalf + kc * 32;
                a.h[2 * v]     = (__bf16)Qs[wid][rr * 64 + kk];
                a.h[2 * v + 1] = (__bf16)Qs[wid][rr * 64 + kk + 1];
            }
            const int bkk = bk0 + kc * 32;
            #pragma unroll
            for (int ct = 0; ct < 4; ++ct) {
                const uint4* w4 = (const uint4*)(WTout + (size_t)bkk * 64 + ct * 16);
                AFrag b;
                #pragma unroll
                for (int s = 0; s < 16; ++s)
                    b.h[s] = WTout[(size_t)(((s < 8) ? s : s + 8) + khalf + kc * 32) * 64 + ct * 16 + rr];
                acc[ct] = __builtin_amdgcn_wmma_f32_16x16x32_bf16(
                    false, a.v, false, b.v, (short)0, acc[ct], false, false);
            }
        }
        #pragma unroll
        for (int ct = 0; ct < 4; ++ct) {
            #pragma unroll
            for (int j = 0; j < 8; ++j) {
                int r  = (lane < 16) ? j : j + 8;
                int el = r >> 2, qq = r & 3;
                int c  = ct * 16 + (lane & 15);
                int dr = ei[(size_t)E_ + e0 + el];
                float v = acc[ct][j] + bout[c];
                atomicAdd(&outbuf[(size_t)dr * CC + qq * 64 + c], v);
            }
        }
    }
}

// ---------------------------------------------------------------------------
extern "C" void kernel_launch(void* const* d_in, const int* in_sizes, int n_in,
                              void* d_out, int out_size, void* d_ws, size_t ws_size,
                              hipStream_t stream)
{
    const float* x_a   = (const float*)d_in[0];
    const float* x_b   = (const float*)d_in[1];
    const int*   ei1   = (const int*)  d_in[2];
    const float* ea1   = (const float*)d_in[3];
    const int*   ei2   = (const int*)  d_in[4];
    const float* ea2   = (const float*)d_in[5];
    const float* Wn_a  = (const float*)d_in[6];
    const float* bn_a  = (const float*)d_in[7];
    const float* Wn_b  = (const float*)d_in[8];
    const float* bn_b  = (const float*)d_in[9];
    const float* We1   = (const float*)d_in[10];
    const float* be1   = (const float*)d_in[11];
    const float* Win1  = (const float*)d_in[12];
    const float* bin1  = (const float*)d_in[13];
    const float* Wout1 = (const float*)d_in[14];
    const float* bout1 = (const float*)d_in[15];
    const float* We2   = (const float*)d_in[16];
    const float* be2   = (const float*)d_in[17];
    const float* Win2  = (const float*)d_in[18];
    const float* bin2  = (const float*)d_in[19];
    const float* Wout2 = (const float*)d_in[20];
    const float* bout2 = (const float*)d_in[21];
    const float* Wo_a  = (const float*)d_in[22];
    const float* bo_a  = (const float*)d_in[23];
    const float* Wo_b  = (const float*)d_in[24];
    const float* bo_b  = (const float*)d_in[25];
    const float* ln_g  = (const float*)d_in[26];
    const float* ln_b  = (const float*)d_in[27];

    const int N = in_sizes[0] / CC;    // 65536
    const int E = in_sizes[3];         // 131072

    // ---- workspace layout ----
    __bf16* wsb = (__bf16*)d_ws;
    __bf16* WTn_a  = wsb;                     // 65536
    __bf16* WTn_b  = WTn_a  + 65536;
    __bf16* WTo_a  = WTn_b  + 65536;
    __bf16* WTo_b  = WTo_a  + 65536;
    __bf16* WTin1  = WTo_b  + 65536;          // 12288
    __bf16* WTin2  = WTin1  + 12288;
    __bf16* WTout1 = WTin2  + 12288;          // 4096
    __bf16* WTout2 = WTout1 + 4096;
    __bf16* xp_a16 = WTout2 + 4096;           // N*256 bf16
    __bf16* xp_b16 = xp_a16 + (size_t)N * CC;
    float*  out_a  = (float*)(xp_b16 + (size_t)N * CC);
    float*  out_b  = out_a + (size_t)N * CC;

    // ---- weight prep (bf16 transposed) ----
    transpose_cvt<<<256, 256, 0, stream>>>(Wn_a,  WTn_a,  256, 256);
    transpose_cvt<<<256, 256, 0, stream>>>(Wn_b,  WTn_b,  256, 256);
    transpose_cvt<<<256, 256, 0, stream>>>(Wo_a,  WTo_a,  256, 256);
    transpose_cvt<<<256, 256, 0, stream>>>(Wo_b,  WTo_b,  256, 256);
    transpose_cvt<<<48,  256, 0, stream>>>(Win1,  WTin1,  192, 64);
    transpose_cvt<<<48,  256, 0, stream>>>(Win2,  WTin2,  192, 64);
    transpose_cvt<<<16,  256, 0, stream>>>(Wout1, WTout1, 64,  64);
    transpose_cvt<<<16,  256, 0, stream>>>(Wout2, WTout2, 64,  64);

    hipMemsetAsync(out_a, 0, 2 * (size_t)N * CC * sizeof(float), stream);

    // ---- node projections -> bf16 xp ----
    gemm256<true, false><<<N / 32, 256, 0, stream>>>(
        x_a, WTn_a, bn_a, nullptr, xp_a16, nullptr, nullptr, N);
    gemm256<true, false><<<N / 32, 256, 0, stream>>>(
        x_b, WTn_b, bn_b, nullptr, xp_b16, nullptr, nullptr, N);

    // ---- edge messages + fused scatter (m1 -> out_b, m2 -> out_a) ----
    edge_mha_scatter<<<E / 16, 128, 0, stream>>>(
        xp_a16, xp_b16, ei1, ea1, We1, be1, WTin1, bin1, WTout1, bout1, out_b, E);
    edge_mha_scatter<<<E / 16, 128, 0, stream>>>(
        xp_b16, xp_a16, ei2, ea2, We2, be2, WTin2, bin2, WTout2, bout2, out_a, E);

    // ---- output projection with fused LayerNorm -> [2,N,C] ----
    float* y = (float*)d_out;
    gemm256<false, true><<<N / 32, 256, 0, stream>>>(
        out_a, WTo_a, bo_a, y, nullptr, ln_g, ln_b, N);
    gemm256<false, true><<<N / 32, 256, 0, stream>>>(
        out_b, WTo_b, bo_b, y + (size_t)N * CC, nullptr, ln_g, ln_b, N);
}
